// GAT_55619826483223
// MI455X (gfx1250) — matrix operand order repaired and checked by reading, not compiled, not run on hardware
//
#include <hip/hip_runtime.h>
#include <hip/hip_fp16.h>

// ---------------- CDNA5 WMMA types ----------------
typedef __attribute__((ext_vector_type(16))) _Float16 v16h;
typedef __attribute__((ext_vector_type(8)))  float    v8f;

union V16H { uint4 u4[2]; v16h v; };

// ---------------- generic f16 WMMA GEMM ----------------
// C[m, cColOff + n] = sum_k A'[m,k] * B[n,k]   (B row-major [Nout,K] == W)
// A'[m,:] = A[rowIdx ? rowIdx[m*idxStride+idxOff] : m, :]
// One wave per 16x16 tile of C; block = (32,4) waves covering 4 n-tiles.
__global__ __launch_bounds__(128)
void gemm_wmma_f16(const _Float16* __restrict__ A, int lda,
                   const int* __restrict__ rowIdx, int idxStride, int idxOff,
                   const _Float16* __restrict__ B,
                   float* __restrict__ C, int ldc, int cColOff,
                   int M, int K)
{
    const int lane = threadIdx.x;        // 0..31
    const int half = lane >> 4;          // 0/1
    const int r    = lane & 15;
    const int mTile = blockIdx.x;
    const int nTile = blockIdx.y * 4 + threadIdx.y;

    int row16 = mTile * 16 + r;
    int arow  = (row16 < M) ? row16 : (M - 1);
    if (rowIdx) arow = rowIdx[(size_t)arow * idxStride + idxOff];
    const int bcol = nTile * 16 + r;

    const _Float16* Abase = A + (size_t)arow * lda;
    const _Float16* Bbase = B + (size_t)bcol * K;

    v8f acc = {};
    for (int k0 = 0; k0 < K; k0 += 32) {
        // A 16x32 f16 fragment: half-wave h holds K {8h..8h+7} in v0-3 and
        // {16+8h..23+8h} in v4-7 -> two 16B loads.
        V16H a, b;
        a.u4[0] = *(const uint4*)(Abase + k0 + 8 * half);
        a.u4[1] = *(const uint4*)(Abase + k0 + 16 + 8 * half);
        // B 32x16 f16 fragment: same K order as A.
        b.u4[0] = *(const uint4*)(Bbase + k0 + 8 * half);
        b.u4[1] = *(const uint4*)(Bbase + k0 + 16 + 8 * half);
        acc = __builtin_amdgcn_wmma_f32_16x16x32_f16(
                false, a.v, false, b.v, (short)0, acc, false, false);
    }

    // C/D layout: lanes 0-15 -> M = 0..7 in v0..7, lanes 16-31 -> M = 8..15.
    float* Cc = C + cColOff + nTile * 16 + r;
    const int mbase = mTile * 16 + 8 * half;
    #pragma unroll
    for (int i = 0; i < 8; ++i) {
        int m = mbase + i;
        if (m < M) Cc[(size_t)m * ldc] = acc[i];
    }
}

// ---------------- helpers ----------------
__device__ __forceinline__ unsigned enc_f32(float f) {
    unsigned u = __float_as_uint(f);
    return u ^ ((unsigned)((int)u >> 31) | 0x80000000u);
}
__device__ __forceinline__ float dec_f32(unsigned k) {
    unsigned u = k ^ ((unsigned)((int)(~k) >> 31) | 0x80000000u);
    return __uint_as_float(u);
}

__global__ void zero_kernel(unsigned* __restrict__ p, size_t n) {
    size_t i = (size_t)blockIdx.x * blockDim.x + threadIdx.x;
    size_t stride = (size_t)gridDim.x * blockDim.x;
    for (; i < n; i += stride) p[i] = 0u;
}

__global__ void cvt_f16_kernel(const float* __restrict__ in,
                               _Float16* __restrict__ out, size_t n) {
    size_t i = (size_t)blockIdx.x * blockDim.x + threadIdx.x;
    size_t stride = (size_t)gridDim.x * blockDim.x;
    for (; i < n; i += stride) out[i] = (_Float16)in[i];
}

__global__ void elu_cvt_kernel(const float* __restrict__ in,
                               _Float16* __restrict__ out, size_t n) {
    size_t i = (size_t)blockIdx.x * blockDim.x + threadIdx.x;
    size_t stride = (size_t)gridDim.x * blockDim.x;
    for (; i < n; i += stride) {
        float x = in[i];
        x = (x > 0.f) ? x : (__expf(x) - 1.f);
        out[i] = (_Float16)x;
    }
}

// s_src[n,hd] = z[n,hd*64:..] . attn[hd,0:64] ; s_dst with attn[hd,64:128]
__global__ void attn_score_kernel(const float* __restrict__ z,
                                  const float* __restrict__ attn,
                                  float* __restrict__ s_src,
                                  float* __restrict__ s_dst,
                                  int n, int heads) {
    int t = blockIdx.x * blockDim.x + threadIdx.x;
    if (t >= n * heads) return;
    int node = t / heads, hd = t - node * heads;
    const float* zr = z + (size_t)node * heads * 64 + hd * 64;
    const float* aw = attn + hd * 128;
    float ss = 0.f, sd = 0.f;
    #pragma unroll
    for (int i = 0; i < 64; ++i) {
        float v = zr[i];
        ss += v * aw[i];
        sd += v * aw[64 + i];
    }
    s_src[t] = ss;
    s_dst[t] = sd;
}

// pass A: e = leaky_relu(s_src[src]+s_dst[dst]); store e; segment max via uint key
__global__ void edge_scores_kernel(const int* __restrict__ src,
                                   const int* __restrict__ dst,
                                   const float* __restrict__ s_src,
                                   const float* __restrict__ s_dst,
                                   float* __restrict__ e_out,
                                   unsigned* __restrict__ mkey,
                                   int E, int heads) {
    int eid = blockIdx.x * blockDim.x + threadIdx.x;
    if (eid >= E) return;
    int s = src[eid], d = dst[eid];
    for (int hd = 0; hd < heads; ++hd) {
        float e = s_src[(size_t)s * heads + hd] + s_dst[(size_t)d * heads + hd];
        e = (e > 0.f) ? e : 0.01f * e;
        e_out[(size_t)eid * heads + hd] = e;
        atomicMax(&mkey[(size_t)d * heads + hd], enc_f32(e));
    }
}

// pass B: ex = exp(e - m[dst]); denom[dst] += ex; e buffer overwritten with ex
__global__ void edge_softmax_kernel(const int* __restrict__ dst,
                                    const unsigned* __restrict__ mkey,
                                    float* __restrict__ e_inout,
                                    float* __restrict__ denom,
                                    int E, int heads) {
    int eid = blockIdx.x * blockDim.x + threadIdx.x;
    if (eid >= E) return;
    int d = dst[eid];
    for (int hd = 0; hd < heads; ++hd) {
        float m = dec_f32(mkey[(size_t)d * heads + hd]);
        float ex = __expf(e_inout[(size_t)eid * heads + hd] - m);
        e_inout[(size_t)eid * heads + hd] = ex;
        atomicAdd(&denom[(size_t)d * heads + hd], ex);
    }
}

// pass C: out[dst, c] += (ex/denom[dst]) * z[src, c]; one block per edge,
// blockDim = heads*64 channels.
__global__ void edge_aggregate_kernel(const int* __restrict__ src,
                                      const int* __restrict__ dst,
                                      const float* __restrict__ ex,
                                      const float* __restrict__ denom,
                                      const float* __restrict__ z,
                                      float* __restrict__ out,
                                      int heads) {
    int eid = blockIdx.x;
    int c = threadIdx.x;
    int hd = c >> 6;
    int s = src[eid], d = dst[eid];
    float alpha = ex[(size_t)eid * heads + hd] / denom[(size_t)d * heads + hd];
    atomicAdd(&out[(size_t)d * heads * 64 + c],
              alpha * z[(size_t)s * heads * 64 + c]);
}

// cat([a(64), b(64)]) -> f16 [n,128]
__global__ void concat2_cvt_kernel(const float* __restrict__ a,
                                   const float* __restrict__ b,
                                   _Float16* __restrict__ out, int n) {
    int t = blockIdx.x * blockDim.x + threadIdx.x;
    if (t >= n * 128) return;
    int row = t >> 7, c = t & 127;
    float v = (c < 64) ? a[(size_t)row * 64 + c] : b[(size_t)row * 64 + (c - 64)];
    out[t] = (_Float16)v;
}

// column sums of x[n,64] accumulated into out[64] (atomicAdd); coalesced reads.
__global__ void mean_reduce_kernel(const float* __restrict__ x,
                                   float* __restrict__ out, int n) {
    __shared__ float sm[256];
    int tid = threadIdx.x;
    int gt = blockIdx.x * 256 + tid;
    int c = tid & 63;
    int stride = (gridDim.x * 256) >> 6;
    float acc = 0.f;
    for (int r = gt >> 6; r < n; r += stride)
        acc += x[(size_t)r * 64 + c];
    sm[tid] = acc;
    __syncthreads();
    if (tid < 64)
        atomicAdd(&out[tid], sm[tid] + sm[tid + 64] + sm[tid + 128] + sm[tid + 192]);
}

__global__ void scale_kernel(float* __restrict__ p, float s, int n) {
    int t = blockIdx.x * blockDim.x + threadIdx.x;
    if (t < n) p[t] *= s;
}

// ---------------- launcher ----------------
static inline size_t align256(size_t x) { return (x + 255) & ~(size_t)255; }

extern "C" void kernel_launch(void* const* d_in, const int* in_sizes, int n_in,
                              void* d_out, int out_size, void* d_ws, size_t ws_size,
                              hipStream_t stream) {
    (void)n_in; (void)out_size; (void)ws_size;
    const float* h       = (const float*)d_in[0];
    const int*   src     = (const int*)  d_in[1];
    const int*   dst     = (const int*)  d_in[2];
    const int*   fcover  = (const int*)  d_in[3];
    const float* l1fc    = (const float*)d_in[4];   // [4,64,128]
    const float* l1attn  = (const float*)d_in[5];   // [4,128]
    const float* l2fc    = (const float*)d_in[6];   // [64,256]
    const float* l2attn  = (const float*)d_in[7];   // [128]
    const float* fracw   = (const float*)d_in[8];   // [3,64,128]
    const float* fracfin = (const float*)d_in[9];   // [64,192]
    const float* fc2w    = (const float*)d_in[10];  // [64,128]

    const int N = in_sizes[0] / 128;
    const int E = in_sizes[1];
    const int HEADS = 4;

    // ---- workspace layout (bump allocator, regions reused over time) ----
    char* ws = (char*)d_ws;
    size_t off = 0;
    auto take = [&](size_t bytes) -> char* {
        char* p = ws + off;
        off = align256(off + bytes);
        return p;
    };
    _Float16* h16      = (_Float16*)take((size_t)N * 128 * 2);
    _Float16* w1all16  = (_Float16*)take((size_t)256 * 128 * 2);
    _Float16* l2fc16   = (_Float16*)take((size_t)64 * 256 * 2);
    _Float16* fracw16  = (_Float16*)take((size_t)3 * 64 * 128 * 2);
    _Float16* fracfin16= (_Float16*)take((size_t)64 * 192 * 2);
    _Float16* fc2w16   = (_Float16*)take((size_t)64 * 128 * 2);
    float*    Z1       = (float*)   take((size_t)N * 256 * 4);  // reused: hnei1_f16, frac16
    float*    AGG      = (float*)   take((size_t)N * 256 * 4);  // reused: frac f32, cat2_f16, hglob
    float*    Ssrc     = (float*)   take((size_t)N * HEADS * 4);
    float*    Sdst     = (float*)   take((size_t)N * HEADS * 4);
    unsigned* MKEY     = (unsigned*)take((size_t)N * HEADS * 4);
    float*    DEN      = (float*)   take((size_t)N * HEADS * 4);
    float*    EX       = (float*)   take((size_t)E * HEADS * 4); // reused: h_frac
    float*    Z2       = (float*)   take((size_t)N * 64 * 4);
    float*    HN2      = (float*)   take((size_t)N * 64 * 4);

    _Float16* hnei1_16 = (_Float16*)Z1;                              // after Z1 dead
    _Float16* frac16   = (_Float16*)Z1;                              // after hnei1 dead
    float*    HFRAC    = EX;                                         // after layer2 done
    _Float16* cat2_16  = (_Float16*)AGG;                             // after frac f32 dead
    float*    HGLOB    = (float*)((char*)AGG + align256((size_t)N * 128 * 2));
    float*    outf     = (float*)d_out;

    const int mTiles = (N + 15) / 16;
    dim3 wblk(32, 4);
    auto gblocks = [](size_t n) { return (unsigned)((n + 255) / 256); };

    // 1) f16 conversions
    cvt_f16_kernel<<<gblocks((size_t)N * 128), 256, 0, stream>>>(h, h16, (size_t)N * 128);
    cvt_f16_kernel<<<gblocks(256 * 128), 256, 0, stream>>>(l1fc, w1all16, 256 * 128);
    cvt_f16_kernel<<<gblocks(64 * 256), 256, 0, stream>>>(l2fc, l2fc16, 64 * 256);
    cvt_f16_kernel<<<gblocks(3 * 64 * 128), 256, 0, stream>>>(fracw, fracw16, 3 * 64 * 128);
    cvt_f16_kernel<<<gblocks(64 * 192), 256, 0, stream>>>(fracfin, fracfin16, 64 * 192);
    cvt_f16_kernel<<<gblocks(64 * 128), 256, 0, stream>>>(fc2w, fc2w16, 64 * 128);

    // 2) layer1 fc: Z1[N,256] = h @ W1all^T  (heads stacked head-major)
    gemm_wmma_f16<<<dim3(mTiles, 4), wblk, 0, stream>>>(
        h16, 128, nullptr, 0, 0, w1all16, Z1, 256, 0, N, 128);

    // 3) layer1 GAT
    attn_score_kernel<<<gblocks((size_t)N * HEADS), 256, 0, stream>>>(
        Z1, l1attn, Ssrc, Sdst, N, HEADS);
    zero_kernel<<<gblocks((size_t)N * HEADS), 256, 0, stream>>>(MKEY, (size_t)N * HEADS);
    edge_scores_kernel<<<gblocks(E), 256, 0, stream>>>(
        src, dst, Ssrc, Sdst, EX, MKEY, E, HEADS);
    zero_kernel<<<gblocks((size_t)N * HEADS), 256, 0, stream>>>(
        (unsigned*)DEN, (size_t)N * HEADS);
    edge_softmax_kernel<<<gblocks(E), 256, 0, stream>>>(dst, MKEY, EX, DEN, E, HEADS);
    zero_kernel<<<gblocks((size_t)N * 256), 256, 0, stream>>>(
        (unsigned*)AGG, (size_t)N * 256);
    edge_aggregate_kernel<<<dim3(E), 256, 0, stream>>>(src, dst, EX, DEN, Z1, AGG, HEADS);

    // 4) ELU + cvt -> hnei1_16 (overwrites Z1 region; Z1 f32 dead)
    elu_cvt_kernel<<<gblocks((size_t)N * 256), 256, 0, stream>>>(
        AGG, hnei1_16, (size_t)N * 256);

    // 5) layer2 fc: Z2[N,64] = hnei1 @ l2fc^T
    gemm_wmma_f16<<<dim3(mTiles, 1), wblk, 0, stream>>>(
        hnei1_16, 256, nullptr, 0, 0, l2fc16, Z2, 64, 0, N, 256);

    // 6) layer2 GAT (heads=1)
    attn_score_kernel<<<gblocks(N), 256, 0, stream>>>(Z2, l2attn, Ssrc, Sdst, N, 1);
    zero_kernel<<<gblocks(N), 256, 0, stream>>>(MKEY, (size_t)N);
    edge_scores_kernel<<<gblocks(E), 256, 0, stream>>>(src, dst, Ssrc, Sdst, EX, MKEY, E, 1);
    zero_kernel<<<gblocks(N), 256, 0, stream>>>((unsigned*)DEN, (size_t)N);
    edge_softmax_kernel<<<gblocks(E), 256, 0, stream>>>(dst, MKEY, EX, DEN, E, 1);
    zero_kernel<<<gblocks((size_t)N * 64), 256, 0, stream>>>(
        (unsigned*)HN2, (size_t)N * 64);
    edge_aggregate_kernel<<<dim3(E), 64, 0, stream>>>(src, dst, EX, DEN, Z2, HN2, 1);

    // 7) fractal branch: AGG reused as f32 [N,192]
    for (int s = 0; s < 3; ++s) {
        gemm_wmma_f16<<<dim3(mTiles, 1), wblk, 0, stream>>>(
            h16, 128, fcover, 3, s, fracw16 + (size_t)s * 64 * 128,
            AGG, 192, s * 64, N, 128);
    }
    cvt_f16_kernel<<<gblocks((size_t)N * 192), 256, 0, stream>>>(
        AGG, frac16, (size_t)N * 192);
    gemm_wmma_f16<<<dim3(mTiles, 1), wblk, 0, stream>>>(
        frac16, 192, nullptr, 0, 0, fracfin16, HFRAC, 64, 0, N, 192);

    // 8) fc2 on cat([HN2, HFRAC])
    concat2_cvt_kernel<<<gblocks((size_t)N * 128), 256, 0, stream>>>(
        HN2, HFRAC, cat2_16, N);
    gemm_wmma_f16<<<dim3(mTiles, 1), wblk, 0, stream>>>(
        cat2_16, 128, nullptr, 0, 0, fc2w16, HGLOB, 64, 0, N, 128);

    // 9) mean over nodes -> d_out[64]
    zero_kernel<<<1, 64, 0, stream>>>((unsigned*)outf, 64);
    mean_reduce_kernel<<<256, 256, 0, stream>>>(HGLOB, outf, N);
    scale_kernel<<<1, 64, 0, stream>>>(outf, 1.0f / (float)N, 64);
}
